// IntraModalityMamba_56281251446875
// MI455X (gfx1250) — matrix operand order repaired and checked by reading, not compiled, not run on hardware
//
#include <hip/hip_runtime.h>
#include <hip/hip_bf16.h>
#include <math.h>

// ---------------------------------------------------------------------------
// Mamba (4 layers) forward for MI455X / gfx1250, wave32.
//   GEMMs: bf16 operands in global -> double-buffered TDM (tensor_load_to_lds)
//          -> ds_load_b128 fragments -> v_wmma_f32_16x16x32_bf16 (f32 accum)
//   Scan : one wave per (b, d); lane = state index n; shfl-xor reduction
// Input flattening (setup_inputs dict order):
//   d_in[0] = x (8,1024,512)
//   per layer l (11 tensors): 1+11*l + j, j in:
//     0 in_proj_w(2048,512) 1 conv_w(1024,1,4) 2 conv_b(1024) 3 x_proj_w(96,1024)
//     4 dt_proj_w(1024,32)  5 dt_proj_b(1024)  6 A_log(1024,32) 7 D(1024)
//     8 out_proj_w(512,1024) 9 ln_g(512) 10 ln_b(512)
// ---------------------------------------------------------------------------

#define BATCH   8
#define SEQ     1024
#define DMODEL  512
#define DINNER  1024
#define DSTATE  32
#define DTRANK  32
#define XPROJ_N (DTRANK + 2 * DSTATE)   // 96
#define BL      (BATCH * SEQ)           // 8192

typedef __attribute__((ext_vector_type(16))) __bf16        bf16x16;
typedef __attribute__((ext_vector_type(8)))  unsigned int  uint8v;
typedef __attribute__((ext_vector_type(8)))  float         f32x8;
typedef __attribute__((ext_vector_type(4)))  unsigned int  u32x4;
typedef __attribute__((ext_vector_type(8)))  int           i32x8;
typedef __attribute__((ext_vector_type(4)))  int           i32x4;

// ---- helpers --------------------------------------------------------------

__device__ __forceinline__ unsigned short f32_to_bf16(float f) {
    unsigned u = __float_as_uint(f);
    unsigned r = u + 0x7FFFu + ((u >> 16) & 1u);   // round-to-nearest-even
    return (unsigned short)(r >> 16);
}
__device__ __forceinline__ float softplus_f(float x) {
    return (x > 20.f) ? x : log1pf(__expf(x));
}
__device__ __forceinline__ float silu_f(float x) {
    return x / (1.f + __expf(-x));
}

// ---- TDM 2-D tile load: global bf16 tensor -> LDS, padded rows ------------
// D# per CDNA5 ISA. data_size=1 (2 bytes).
// pad_enable=1, pad_interval=3 (16 dwords = one 32-elem bf16 row),
// pad_amount=3 (4 dwords)  =>  LDS row stride = 20 dwords (80 bytes),
// keeping every row 16-byte aligned so fragments are ds_load_b128-able.
// clang-23/therock 6-arg builtin: (u32x4, i32x8, i32x4, i32x4, i32x8, i32).
__device__ __forceinline__ void tdm_load_tile_bf16(
        unsigned lds_byte_addr,          // LDS byte offset of destination
        const unsigned short* gsrc,      // global address of tile start
        unsigned tensor_d0,              // row length available (elems)
        unsigned tensor_d1,              // rows available (OOB rows -> zeros)
        unsigned tile_d0,                // tile width  (elems)  (mult of 2)
        unsigned tile_d1,                // tile height (rows)
        unsigned long long stride0)      // row stride (elems)
{
    unsigned long long ga = (unsigned long long)gsrc;
    u32x4 g0;
    g0[0] = 1u;                                          // count=1, user mode
    g0[1] = lds_byte_addr;
    g0[2] = (unsigned)(ga & 0xffffffffu);
    g0[3] = (unsigned)((ga >> 32) & 0x01ffffffu) | (2u << 30);   // type=2
    i32x8 g1;
    g1[0] = (int)((1u << 16) | (1u << 20) | (3u << 22) | (3u << 25));
    //             dsz=2B      pad_en      ivl=16dw      amount=4dw
    g1[1] = (int)((tensor_d0 & 0xffffu) << 16);
    g1[2] = (int)(((tensor_d0 >> 16) & 0xffffu) | ((tensor_d1 & 0xffffu) << 16));
    g1[3] = (int)(((tensor_d1 >> 16) & 0xffffu) | ((tile_d0 & 0xffffu) << 16));
    g1[4] = (int)(tile_d1 & 0xffffu);                    // tile_dim2 = 0
    g1[5] = (int)(unsigned)(stride0 & 0xffffffffu);
    g1[6] = (int)(unsigned)((stride0 >> 32) & 0xffffu);  // dim1_stride = 0
    g1[7] = 0;
    i32x4 g2 = {0, 0, 0, 0};                             // 2-D: groups 2/3 unused
    i32x4 g3 = {0, 0, 0, 0};
    i32x8 g4 = {0, 0, 0, 0, 0, 0, 0, 0};
    __builtin_amdgcn_tensor_load_to_lds(g0, g1, g2, g3, g4, 0);
}

// ---- GEMM: C[M,N] = A[M,K] @ W[N,K]^T  (bf16 in, f32 out, WMMA) -----------
// 256 threads = 8 waves; tile 128x64; wave w owns rows m0+16w..m0+16w+15.
// Double-buffered TDM staging: stage s+1 DMA overlaps stage s compute.

#define TM 128
#define TN 64
#define TK 32
#define LDS_STRIDE 20            // dwords per LDS row (16 data + 4 TDM pad)
#define A_TILE_DW (TM * LDS_STRIDE)
#define W_TILE_DW (TN * LDS_STRIDE)

__global__ void __launch_bounds__(256)
gemm_wmma_tdm(const unsigned short* __restrict__ A, int lda,
              const unsigned short* __restrict__ W,   // [N][K] row-major bf16
              const float* __restrict__ bias,         // nullable
              float* __restrict__ C,
              unsigned short* __restrict__ Cbf,       // nullable bf16 mirror
              int ldc, int M, int N, int K, int mode) // mode&1 softplus+bias
{                                                     // mode&2 also store bf16
    __shared__ unsigned int sA[2 * A_TILE_DW];
    __shared__ unsigned int sW[2 * W_TILE_DW];

    const int tid  = threadIdx.x;
    const int lane = tid & 31;
    const int wave = tid >> 5;              // 0..7
    const int m0   = blockIdx.y * TM;
    const int n0   = blockIdx.x * TN;
    const int g    = lane >> 4;             // lane group (ISA fragment halves)
    const int lm   = lane & 15;

    const unsigned ldsA = (unsigned)(unsigned long long)(&sA[0]);
    const unsigned ldsW = (unsigned)(unsigned long long)(&sW[0]);

    f32x8 acc[4];
    #pragma unroll
    for (int nt = 0; nt < 4; ++nt)
        #pragma unroll
        for (int i = 0; i < 8; ++i) acc[nt][i] = 0.f;

    const int S = K / TK;                   // number of K stages (K % 32 == 0)

    // prologue: stage 0 into buffer 0
    if (wave == 0) {
        tdm_load_tile_bf16(ldsA, A + (size_t)m0 * lda, (unsigned)lda,
                           (unsigned)(M - m0), TK, TM, (unsigned long long)lda);
        tdm_load_tile_bf16(ldsW, W + (size_t)n0 * K, (unsigned)K,
                           (unsigned)(N - n0), TK, TN, (unsigned long long)K);
    }

    for (int s = 0; s < S; ++s) {
        const int cb = s & 1;               // compute buffer
        if (wave == 0) {
            if (s + 1 < S) {                // issue next stage, then retire s
                const int nb = (s + 1) & 1;
                const int k1 = (s + 1) * TK;
                tdm_load_tile_bf16(ldsA + nb * (A_TILE_DW * 4),
                                   A + (size_t)m0 * lda + k1,
                                   (unsigned)(lda - k1), (unsigned)(M - m0),
                                   TK, TM, (unsigned long long)lda);
                tdm_load_tile_bf16(ldsW + nb * (W_TILE_DW * 4),
                                   W + (size_t)n0 * K + k1,
                                   (unsigned)(K - k1), (unsigned)(N - n0),
                                   TK, TN, (unsigned long long)K);
                __builtin_amdgcn_s_wait_tensorcnt(2);   // stage s complete
            } else {
                __builtin_amdgcn_s_wait_tensorcnt(0);
            }
        }
        __syncthreads();                    // publish stage s tiles

        const unsigned int* tA = sA + cb * A_TILE_DW;
        const unsigned int* tW = sW + cb * W_TILE_DW;

        // -- A fragment (ISA 16-bit A 16x32 layout) -------------------------
        uint8v aw;
        {
            int m = (wave << 4) + lm;
            #pragma unroll
            for (int r = 0; r < 8; ++r) {
                int dw = (r & 3) + 4 * g + ((r >> 2) << 3);  // K-pair dword idx
                aw[r] = tA[m * LDS_STRIDE + dw];
            }
        }
        bf16x16 afrag = __builtin_bit_cast(bf16x16, aw);

        // -- 4 B fragments + WMMA ------------------------------------------
        #pragma unroll
        for (int nt = 0; nt < 4; ++nt) {
            uint8v bw;
            int n = (nt << 4) + lm;                          // local W row
            #pragma unroll
            for (int r = 0; r < 8; ++r)
                bw[r] = tW[n * LDS_STRIDE + (r & 3) + 4 * g + ((r >> 2) << 3)];  // K-pair dword idx (same as A)
            bf16x16 bfrag = __builtin_bit_cast(bf16x16, bw);
            acc[nt] = __builtin_amdgcn_wmma_f32_16x16x32_bf16(
                          false, afrag, false, bfrag,
                          (short)0, acc[nt], false, false);
        }
        __syncthreads();                    // stage s reads done before overwrite
    }

    // -- epilogue + store (C layout: lane = N, vgpr r -> M = r + 8g) --------
    #pragma unroll
    for (int nt = 0; nt < 4; ++nt) {
        int n = n0 + (nt << 4) + lm;
        if (n >= N) continue;
        float bv = (mode & 1) ? bias[n] : 0.f;
        #pragma unroll
        for (int r = 0; r < 8; ++r) {
            int m = m0 + (wave << 4) + r + 8 * g;
            float v = acc[nt][r];
            if (mode & 1) v = softplus_f(v + bv);
            C[(size_t)m * ldc + n] = v;
            if (mode & 2) Cbf[(size_t)m * ldc + n] = f32_to_bf16(v);
        }
    }
}

// ---- f32 -> bf16 elementwise convert --------------------------------------
__global__ void __launch_bounds__(256)
cvt_bf16_kernel(const float* __restrict__ src, unsigned short* __restrict__ dst,
                int n)
{
    int i = blockIdx.x * 256 + threadIdx.x;
    if (i < n) dst[i] = f32_to_bf16(src[i]);
}

// ---- depthwise causal conv (k=4) + bias + SiLU ----------------------------
__global__ void __launch_bounds__(256)
conv_silu_kernel(const float* __restrict__ xz,   // [BL][2048]; xm_pre = cols 0..1023
                 const float* __restrict__ cw,   // [1024][4]
                 const float* __restrict__ cb,   // [1024]
                 float* __restrict__ xm,         // [BL][1024]
                 unsigned short* __restrict__ xm_bf)
{
    int i = blockIdx.x * 256 + threadIdx.x;      // over B*L*DINNER
    int d = i & (DINNER - 1);
    int t = (i >> 10) & (SEQ - 1);
    int b = i >> 20;
    float w0 = cw[d * 4 + 0], w1 = cw[d * 4 + 1];
    float w2 = cw[d * 4 + 2], w3 = cw[d * 4 + 3];
    size_t rb = ((size_t)b * SEQ + t) * 2048 + d;
    float acc = cb[d];
    if (t >= 3) acc += w0 * xz[rb - 3 * 2048];
    if (t >= 2) acc += w1 * xz[rb - 2 * 2048];
    if (t >= 1) acc += w2 * xz[rb - 1 * 2048];
    acc += w3 * xz[rb];
    acc = silu_f(acc);
    size_t o = ((size_t)b * SEQ + t) * DINNER + d;
    xm[o] = acc;
    xm_bf[o] = f32_to_bf16(acc);
}

// ---- selective scan: one wave per (b,d); lane = state n -------------------
__global__ void __launch_bounds__(256)
scan_kernel(const float* __restrict__ xm,    // u     [BL][1024]
            const float* __restrict__ dl,    // delta [BL][1024]
            const float* __restrict__ xdbl,  // [BL][96] : dt|B|C
            const float* __restrict__ xz,    // z at cols 1024..2047
            const float* __restrict__ A_log, // [1024][32]
            const float* __restrict__ Dp,    // [1024]
            float* __restrict__ y,           // [BL][1024]
            unsigned short* __restrict__ y_bf)
{
    int wid = blockIdx.x * 8 + (threadIdx.x >> 5);   // 0..8191
    int n   = threadIdx.x & 31;
    int b   = wid >> 10;
    int d   = wid & (DINNER - 1);

    float a  = -__expf(A_log[d * DSTATE + n]);
    float Dd = Dp[d];
    float h  = 0.f;

    for (int t = 0; t < SEQ; ++t) {
        size_t row = (size_t)b * SEQ + t;
        float dt = dl[row * DINNER + d];             // wave-uniform (broadcast)
        float u  = xm[row * DINNER + d];
        float Bn = xdbl[row * XPROJ_N + DTRANK + n];
        float Cn = xdbl[row * XPROJ_N + DTRANK + DSTATE + n];
        float dA = __expf(dt * a);
        h = h * dA + (dt * u) * Bn;
        float acc = h * Cn;
        #pragma unroll
        for (int o = 16; o > 0; o >>= 1) acc += __shfl_xor(acc, o, 32);
        if (n == 0) {
            float z  = xz[row * 2048 + DINNER + d];
            float yy = (acc + u * Dd) * silu_f(z);
            y[row * DINNER + d]    = yy;
            y_bf[row * DINNER + d] = f32_to_bf16(yy);
        }
    }
}

// ---- residual add + LayerNorm: one wave per 512-wide row ------------------
__global__ void __launch_bounds__(256)
ln_res_kernel(const float* __restrict__ o,    // [BL][512]  (out_proj result)
              const float* __restrict__ res,  // [BL][512]  (layer input)
              const float* __restrict__ gam,
              const float* __restrict__ bet,
              float* __restrict__ out,        // [BL][512]
              unsigned short* __restrict__ out_bf)
{
    int row  = blockIdx.x * 8 + (threadIdx.x >> 5);
    int lane = threadIdx.x & 31;
    const float* op = o   + (size_t)row * DMODEL;
    const float* rp = res + (size_t)row * DMODEL;
    float v[16];
    float s = 0.f, s2 = 0.f;
    #pragma unroll
    for (int i = 0; i < 16; ++i) {
        int c = lane + 32 * i;
        v[i] = op[c] + rp[c];
        s += v[i];
        s2 += v[i] * v[i];
    }
    #pragma unroll
    for (int w = 16; w > 0; w >>= 1) {
        s  += __shfl_xor(s,  w, 32);
        s2 += __shfl_xor(s2, w, 32);
    }
    float mean = s * (1.f / DMODEL);
    float var  = s2 * (1.f / DMODEL) - mean * mean;
    float inv  = rsqrtf(var + 1e-5f);
    float* outp = out + (size_t)row * DMODEL;
    unsigned short* outbp = out_bf + (size_t)row * DMODEL;
    #pragma unroll
    for (int i = 0; i < 16; ++i) {
        int c = lane + 32 * i;
        float r = (v[i] - mean) * inv * gam[c] + bet[c];
        outp[c]  = r;
        outbp[c] = f32_to_bf16(r);
    }
}

// ---------------------------------------------------------------------------

extern "C" void kernel_launch(void* const* d_in, const int* in_sizes, int n_in,
                              void* d_out, int out_size, void* d_ws, size_t ws_size,
                              hipStream_t stream)
{
    (void)in_sizes; (void)n_in; (void)out_size; (void)ws_size;
    const float* x = (const float*)d_in[0];
    auto P = [&](int l, int j) { return (const float*)d_in[1 + l * 11 + j]; };

    // -- f32 workspace ------------------------------------------------------
    float* ws   = (float*)d_ws;
    float* cur  = ws;                         // [BL][512]  layer output chain
    float* xz   = cur  + (size_t)BL * 512;    // [BL][2048] in_proj out (xm_pre|z)
    float* xm   = xz   + (size_t)BL * 2048;   // [BL][1024] conv+silu
    float* xdbl = xm   + (size_t)BL * 1024;   // [BL][96]   dt|B|C
    float* dl   = xdbl + (size_t)BL * 96;     // [BL][1024] delta
    float* ys   = dl   + (size_t)BL * 1024;   // [BL][1024] scan out (gated)
    float* ob   = ys   + (size_t)BL * 1024;   // [BL][512]  out_proj out
    // -- bf16 workspace (activation mirrors + converted weights) ------------
    unsigned short* bfp   = (unsigned short*)(ob + (size_t)BL * 512);
    unsigned short* xbf   = bfp;                              // [BL][512]
    unsigned short* xm_bf = xbf   + (size_t)BL * 512;         // [BL][1024]
    unsigned short* xd_bf = xm_bf + (size_t)BL * 1024;        // [BL][96]
    unsigned short* ys_bf = xd_bf + (size_t)BL * 96;          // [BL][1024]
    unsigned short* w_in  = ys_bf + (size_t)BL * 1024;        // [2048][512]
    unsigned short* w_xp  = w_in  + (size_t)(2 * DINNER) * DMODEL;  // [96][1024]
    unsigned short* w_dt  = w_xp  + (size_t)XPROJ_N * DINNER;       // [1024][32]
    unsigned short* w_out = w_dt  + (size_t)DINNER * DTRANK;        // [512][1024]

    // bf16 mirror of the first layer's input
    cvt_bf16_kernel<<<(BL * DMODEL) / 256, 256, 0, stream>>>(x, xbf, BL * DMODEL);

    for (int l = 0; l < 4; ++l) {
        const float* Ain = (l == 0) ? x : cur;

        // convert this layer's GEMM weights to bf16
        cvt_bf16_kernel<<<(2 * DINNER * DMODEL) / 256, 256, 0, stream>>>(
            P(l, 0), w_in, 2 * DINNER * DMODEL);
        cvt_bf16_kernel<<<(XPROJ_N * DINNER) / 256, 256, 0, stream>>>(
            P(l, 3), w_xp, XPROJ_N * DINNER);
        cvt_bf16_kernel<<<(DINNER * DTRANK) / 256, 256, 0, stream>>>(
            P(l, 4), w_dt, DINNER * DTRANK);
        cvt_bf16_kernel<<<(DMODEL * DINNER) / 256, 256, 0, stream>>>(
            P(l, 8), w_out, DMODEL * DINNER);

        // xz = Ain @ in_proj_w^T        (8192x512 @ 512x2048)
        gemm_wmma_tdm<<<dim3(2 * DINNER / TN, BL / TM), 256, 0, stream>>>(
            xbf, DMODEL, w_in, nullptr, xz, nullptr, 2 * DINNER,
            BL, 2 * DINNER, DMODEL, 0);

        // xm = silu(depthwise_conv(xm_pre) + conv_b)   (+ bf16 mirror)
        conv_silu_kernel<<<(BL * DINNER) / 256, 256, 0, stream>>>(
            xz, P(l, 1), P(l, 2), xm, xm_bf);

        // xdbl = xm @ x_proj_w^T        (8192x1024 @ 1024x96)  (+ bf16 mirror)
        gemm_wmma_tdm<<<dim3((XPROJ_N + TN - 1) / TN, BL / TM), 256, 0, stream>>>(
            xm_bf, DINNER, w_xp, nullptr, xdbl, xd_bf, XPROJ_N,
            BL, XPROJ_N, DINNER, 2);

        // delta = softplus(dt @ dt_proj_w^T + dt_proj_b)   (K=32, dt=xdbl[:, :32])
        gemm_wmma_tdm<<<dim3(DINNER / TN, BL / TM), 256, 0, stream>>>(
            xd_bf, XPROJ_N, w_dt, P(l, 5), dl, nullptr, DINNER,
            BL, DINNER, DTRANK, 1);

        // selective scan + D skip + z gating   (+ bf16 mirror)
        scan_kernel<<<(BATCH * DINNER) / 8, 256, 0, stream>>>(
            xm, dl, xdbl, xz, P(l, 6), P(l, 7), ys, ys_bf);

        // ob = ys @ out_proj_w^T        (8192x1024 @ 1024x512)
        gemm_wmma_tdm<<<dim3(DMODEL / TN, BL / TM), 256, 0, stream>>>(
            ys_bf, DINNER, w_out, nullptr, ob, nullptr, DMODEL,
            BL, DMODEL, DINNER, 0);

        // next_x = LayerNorm(ob + Ain)   (+ bf16 mirror for next in_proj)
        float* nxt = (l == 3) ? (float*)d_out : cur;
        ln_res_kernel<<<BL / 8, 256, 0, stream>>>(
            ob, Ain, P(l, 9), P(l, 10), nxt, xbf);
    }
}
